// Transformer_76441827934520
// MI455X (gfx1250) — matrix operand order repaired and checked by reading, not compiled, not run on hardware
//
#include <hip/hip_runtime.h>

// ---------------------------------------------------------------------------
// CDNA5 (gfx1250) encoder-decoder transformer forward pass.
// GEMMs: V_WMMA_F32_16X16X32_BF16, wave32, 32x32 wave tiles in a 64x64 block
// tile (128 threads), A/B staged through LDS with double-buffered
// GLOBAL_LOAD_ASYNC_TO_LDS_B128 copies (ASYNCcnt-tracked).
// ---------------------------------------------------------------------------

typedef __attribute__((ext_vector_type(16))) __bf16 v16bf;
typedef __attribute__((ext_vector_type(8)))  float  v8f;
typedef __attribute__((ext_vector_type(4)))  int    v4i;

constexpr int kL = 6, kH = 7, kD = 448, kHS = 64, kV = 32000, kT = 256,
              kFF = 1792, kB = 8;
constexpr int kBT = kB * kT;   // 2048
constexpr int kBH = kB * kH;   // 56
typedef long long ll;

struct BOffs { ll div, s1, mod, s2; };   // offset = (z/div)*s1 + (z%mod)*s2

#if defined(__gfx1250__) && __has_builtin(__builtin_amdgcn_global_load_async_to_lds_b128)
#define HAVE_ASYNC_LDS 1
#else
#define HAVE_ASYNC_LDS 0
#endif

#if HAVE_ASYNC_LDS
typedef __attribute__((address_space(1))) v4i glob_v4i;  // global-AS int4
typedef __attribute__((address_space(3))) v4i lds_v4i;   // LDS-AS int4
#if __has_builtin(__builtin_amdgcn_s_wait_asynccnt)
#define ASYNC_WAIT(n) __builtin_amdgcn_s_wait_asynccnt(n)
#else
#define ASYNC_WAIT(n) asm volatile("s_wait_asynccnt %0" ::"i"(n) : "memory")
#endif
#else
#define ASYNC_WAIT(n) ((void)0)
#endif

// Copy 16 bytes global -> LDS. Async (ASYNCcnt) when available.
__device__ __forceinline__ void cp_async16(float* lds_dst,
                                           const float* glb_src) {
#if HAVE_ASYNC_LDS
  __builtin_amdgcn_global_load_async_to_lds_b128(
      (glob_v4i*)glb_src, (lds_v4i*)lds_dst, 0, 0);
#else
  *(float4*)lds_dst = *(const float4*)glb_src;
#endif
}

__device__ __forceinline__ float4 ld4(const float* p) {
  return *(const float4*)p;
}
__device__ __forceinline__ void cvt4(v16bf& v, int base, float4 x) {
  v[base + 0] = (__bf16)x.x; v[base + 1] = (__bf16)x.y;
  v[base + 2] = (__bf16)x.z; v[base + 3] = (__bf16)x.w;
}

// LDS tile geometry (pads keep 16B alignment: 36*4=144, 68*4=272).
constexpr int kAP  = 36;          // A tile pitch (64 rows x 32 cols)
constexpr int kBPN = 68;          // B tile pitch, normal   (32 rows x 64 cols)
constexpr int kBPT = 36;          // B tile pitch, transB   (64 rows x 32 cols)
constexpr int kASZ = 64 * kAP;    // 2304 floats
constexpr int kBSZ = 64 * kBPT;   // 2304 floats (>= 32*kBPN = 2176)

// C[m,n] = alpha * sum_k A[m,k]*B[k,n] (+bias[n]) (+Res[m,n]) (relu?)
// transB: B element (k,n) read at B[n*ldb + k] (for Q*K^T).
__global__ __launch_bounds__(128) void gemm_bf16_wmma(
    const float* __restrict__ A, const float* __restrict__ Bm,
    const float* __restrict__ bias, const float* Res, float* C,
    int M, int N, int K, int lda, int ldb, int ldc,
    BOffs oa, BOffs ob, BOffs oc, int transB, int relu, float alpha) {
  __shared__ float As[2 * kASZ];
  __shared__ float Bs[2 * kBSZ];

  const int tid  = threadIdx.x;
  const int lane = tid & 31;
  const int wave = tid >> 5;
  const int wm = wave >> 1, wn = wave & 1;
  const ll z = blockIdx.z;
  A  += (z / oa.div) * oa.s1 + (z % oa.mod) * oa.s2;
  Bm += (z / ob.div) * ob.s1 + (z % ob.mod) * ob.s2;
  const ll coff = (z / oc.div) * oc.s1 + (z % oc.mod) * oc.s2;
  C += coff;
  if (Res) Res += coff;

  const int m0blk = blockIdx.y * 64;
  const int n0blk = blockIdx.x * 64;
  const int mlane = lane & 15;     // row (A) / col (B,C) within 16-tile
  const int hl = lane >> 4;        // lane half

  // Cooperative tile copy: 8 x 16B chunks per thread (8 async ops per wave).
  auto issue_copies = [&](int k0, int buf) {
    {  // A tile: 64 rows x 32 floats, thread -> (row, 16-float half)
      const int row = tid >> 1, hf = tid & 1;
      const float* src = A + (ll)(m0blk + row) * lda + k0 + hf * 16;
      float* dst = As + buf * kASZ + row * kAP + hf * 16;
      cp_async16(dst, src);           cp_async16(dst + 4, src + 4);
      cp_async16(dst + 8, src + 8);   cp_async16(dst + 12, src + 12);
    }
    if (!transB) {  // B tile: 32 k-rows x 64 floats, thread -> (row, quarter)
      const int row = tid >> 2, qt = tid & 3;
      const float* src = Bm + (ll)(k0 + row) * ldb + n0blk + qt * 16;
      float* dst = Bs + buf * kBSZ + row * kBPN + qt * 16;
      cp_async16(dst, src);           cp_async16(dst + 4, src + 4);
      cp_async16(dst + 8, src + 8);   cp_async16(dst + 12, src + 12);
    } else {        // B^T tile: 64 n-rows x 32 floats (transpose via indexing)
      const int row = tid >> 1, hf = tid & 1;
      const float* src = Bm + (ll)(n0blk + row) * ldb + k0 + hf * 16;
      float* dst = Bs + buf * kBSZ + row * kBPT + hf * 16;
      cp_async16(dst, src);           cp_async16(dst + 4, src + 4);
      cp_async16(dst + 8, src + 8);   cp_async16(dst + 12, src + 12);
    }
  };

  const int ktiles = K >> 5;
  issue_copies(0, 0);

  v8f acc[2][2] = {};
  for (int kt = 0; kt < ktiles; ++kt) {
    const int cur = kt & 1;
    if (kt + 1 < ktiles) {
      issue_copies((kt + 1) << 5, cur ^ 1);
      ASYNC_WAIT(8);                  // in-order: releases previous 8 chunks
    } else {
      ASYNC_WAIT(0);
    }
    __syncthreads();                  // tile `cur` visible to all waves

    const float* Ab = As + cur * kASZ;
    const float* Bb = Bs + cur * kBSZ;
    v16bf af[2], bfr[2];
    // A fragment: 16x32 bf16. Element i -> k_local = hl*16 + i
#pragma unroll
    for (int mi = 0; mi < 2; ++mi) {
      const float* ap = Ab + (wm * 32 + mi * 16 + mlane) * kAP + hl * 16;
      cvt4(af[mi], 0,  ld4(ap));      cvt4(af[mi], 4,  ld4(ap + 4));
      cvt4(af[mi], 8,  ld4(ap + 8));  cvt4(af[mi], 12, ld4(ap + 12));
    }
    // B fragment: 32x16 bf16. Element i -> k_local = hl*16 + i
#pragma unroll
    for (int ni = 0; ni < 2; ++ni) {
      const int nl = wn * 32 + ni * 16 + mlane;
      if (!transB) {
        const float* bp = Bb + hl * 16 * kBPN + nl;
#pragma unroll
        for (int i = 0; i < 16; ++i) bfr[ni][i] = (__bf16)bp[i * kBPN];
      } else {
        const float* bp = Bb + nl * kBPT + hl * 16;
        cvt4(bfr[ni], 0, ld4(bp));     cvt4(bfr[ni], 4,  ld4(bp + 4));
        cvt4(bfr[ni], 8, ld4(bp + 8)); cvt4(bfr[ni], 12, ld4(bp + 12));
      }
    }
#pragma unroll
    for (int mi = 0; mi < 2; ++mi)
#pragma unroll
      for (int ni = 0; ni < 2; ++ni)
        acc[mi][ni] = __builtin_amdgcn_wmma_f32_16x16x32_bf16(
            false, af[mi], false, bfr[ni], (short)0, acc[mi][ni], false, false);
    __syncthreads();                  // all reads done before buffer reuse
  }

  // C/D layout: VGPR r holds row m = r + 8*hl, column = mlane
#pragma unroll
  for (int mi = 0; mi < 2; ++mi)
#pragma unroll
    for (int ni = 0; ni < 2; ++ni)
#pragma unroll
      for (int r = 0; r < 8; ++r) {
        const int row = m0blk + wm * 32 + mi * 16 + r + hl * 8;
        const int col = n0blk + wn * 32 + ni * 16 + mlane;
        float v = acc[mi][ni][r] * alpha;
        if (bias) v += bias[col];
        const ll idx = (ll)row * ldc + col;
        if (Res) v += Res[idx];
        if (relu) v = fmaxf(v, 0.f);
        C[idx] = v;
      }
}

// ---------------- elementwise / reduction kernels --------------------------

__global__ void embed_kernel(const int* __restrict__ tok,
                             const float* __restrict__ temb,
                             const float* __restrict__ pemb,
                             float* __restrict__ out, int Tlen, int D,
                             ll total) {
  for (ll i = (ll)blockIdx.x * blockDim.x + threadIdx.x; i < total;
       i += (ll)gridDim.x * blockDim.x) {
    const ll d = i % D, bt = i / D, t = bt % Tlen;
    out[i] = temb[(ll)tok[bt] * D + d] + pemb[t * D + d];
  }
}

__global__ __launch_bounds__(128) void ln_kernel(const float* x,
                                                 const float* __restrict__ g,
                                                 const float* __restrict__ b,
                                                 float* y, int D) {
  const ll row = blockIdx.x;
  const float* xr = x + row * D;
  float s = 0.f, ss = 0.f;
  for (int i = threadIdx.x; i < D; i += 128) {
    const float v = xr[i];
    s += v; ss += v * v;
  }
  __shared__ float sh1[128], sh2[128];
  sh1[threadIdx.x] = s; sh2[threadIdx.x] = ss;
  __syncthreads();
  for (int o = 64; o > 0; o >>= 1) {
    if (threadIdx.x < o) {
      sh1[threadIdx.x] += sh1[threadIdx.x + o];
      sh2[threadIdx.x] += sh2[threadIdx.x + o];
    }
    __syncthreads();
  }
  const float mean = sh1[0] / D;
  const float var = sh2[0] / D - mean * mean;
  const float inv = rsqrtf(var + 1e-5f);
  for (int i = threadIdx.x; i < D; i += 128)
    y[row * D + i] = (xr[i] - mean) * inv * g[i] + b[i];
}

// One block (= T threads) per score row; scale already applied by GEMM alpha.
__global__ __launch_bounds__(256) void softmax_mask(float* S,
                                                    const int* __restrict__ km,
                                                    int causal, int Tlen,
                                                    int Hn) {
  const ll rowid = blockIdx.x;
  const int t = (int)(rowid % Tlen);
  const ll z = rowid / Tlen;
  const int b = (int)(z / Hn);
  float* row = S + rowid * Tlen;
  const int s = threadIdx.x;
  const bool valid = (km[b * Tlen + s] != 0) && (!causal || s <= t);
  float v = valid ? row[s] : -3.402823466e38f;
  __shared__ float sh[256];
  sh[s] = v; __syncthreads();
  for (int o = 128; o > 0; o >>= 1) {
    if (s < o) sh[s] = fmaxf(sh[s], sh[s + o]);
    __syncthreads();
  }
  const float m = sh[0];
  __syncthreads();
  const float e = valid ? expf(v - m) : 0.f;
  sh[s] = e; __syncthreads();
  for (int o = 128; o > 0; o >>= 1) {
    if (s < o) sh[s] += sh[s + o];
    __syncthreads();
  }
  row[s] = e / sh[0];
}

__global__ __launch_bounds__(256) void loss_kernel(
    const float* __restrict__ logits, const int* __restrict__ y,
    float* __restrict__ rowloss, int V) {
  const ll row = blockIdx.x;
  const float* lr = logits + row * (ll)V;
  __shared__ float sh[256];
  float m = -3.402823466e38f;
  for (int i = threadIdx.x; i < V; i += 256) m = fmaxf(m, lr[i]);
  sh[threadIdx.x] = m; __syncthreads();
  for (int o = 128; o > 0; o >>= 1) {
    if (threadIdx.x < o)
      sh[threadIdx.x] = fmaxf(sh[threadIdx.x], sh[threadIdx.x + o]);
    __syncthreads();
  }
  m = sh[0];
  __syncthreads();
  float s = 0.f;
  for (int i = threadIdx.x; i < V; i += 256) s += expf(lr[i] - m);
  sh[threadIdx.x] = s; __syncthreads();
  for (int o = 128; o > 0; o >>= 1) {
    if (threadIdx.x < o) sh[threadIdx.x] += sh[threadIdx.x + o];
    __syncthreads();
  }
  if (threadIdx.x == 0)
    rowloss[row] = (m + logf(sh[0])) - lr[y[row]];
}

__global__ __launch_bounds__(256) void reduce_loss(const float* __restrict__ rl,
                                                   float* out, int n) {
  __shared__ float sh[256];
  float s = 0.f;
  for (int i = threadIdx.x; i < n; i += 256) s += rl[i];
  sh[threadIdx.x] = s; __syncthreads();
  for (int o = 128; o > 0; o >>= 1) {
    if (threadIdx.x < o) sh[threadIdx.x] += sh[threadIdx.x + o];
    __syncthreads();
  }
  if (threadIdx.x == 0) *out = sh[0] / n;
}

// ---------------------------------------------------------------------------

extern "C" void kernel_launch(void* const* d_in, const int* in_sizes, int n_in,
                              void* d_out, int out_size, void* d_ws,
                              size_t ws_size, hipStream_t stream) {
  (void)in_sizes; (void)n_in; (void)out_size; (void)ws_size;
  // ---- inputs (setup_inputs dict order, params flattened in insertion order)
  const int*   x     = (const int*)d_in[0];
  const int*   y     = (const int*)d_in[1];
  const int*   mask  = (const int*)d_in[2];
  const int*   tmask = (const int*)d_in[3];
  const float* e_wq  = (const float*)d_in[4];
  const float* e_wk  = (const float*)d_in[5];
  const float* e_wv  = (const float*)d_in[6];
  const float* e_wo  = (const float*)d_in[7];
  const float* e_bo  = (const float*)d_in[8];
  const float* e_w1  = (const float*)d_in[9];
  const float* e_b1  = (const float*)d_in[10];
  const float* e_w2  = (const float*)d_in[11];
  const float* e_b2  = (const float*)d_in[12];
  const float* e_ln1g = (const float*)d_in[13];
  const float* e_ln1b = (const float*)d_in[14];
  const float* e_ln2g = (const float*)d_in[15];
  const float* e_ln2b = (const float*)d_in[16];
  const float* d_sa_wq = (const float*)d_in[17];
  const float* d_sa_wk = (const float*)d_in[18];
  const float* d_sa_wv = (const float*)d_in[19];
  const float* d_sa_wo = (const float*)d_in[20];
  const float* d_sa_bo = (const float*)d_in[21];
  const float* d_ca_wq = (const float*)d_in[22];
  const float* d_ca_wk = (const float*)d_in[23];
  const float* d_ca_wv = (const float*)d_in[24];
  const float* d_ca_wo = (const float*)d_in[25];
  const float* d_ca_bo = (const float*)d_in[26];
  const float* d_w1  = (const float*)d_in[27];
  const float* d_b1  = (const float*)d_in[28];
  const float* d_w2  = (const float*)d_in[29];
  const float* d_b2  = (const float*)d_in[30];
  const float* d_ln1g = (const float*)d_in[31];
  const float* d_ln1b = (const float*)d_in[32];
  const float* d_ln2g = (const float*)d_in[33];
  const float* d_ln2b = (const float*)d_in[34];
  const float* d_ln3g = (const float*)d_in[35];
  const float* d_ln3b = (const float*)d_in[36];
  const float* enc_tok = (const float*)d_in[37];
  const float* enc_pos = (const float*)d_in[38];
  const float* dec_tok = (const float*)d_in[39];
  const float* dec_pos = (const float*)d_in[40];
  const float* enc_fng = (const float*)d_in[41];
  const float* enc_fnb = (const float*)d_in[42];
  const float* dec_fng = (const float*)d_in[43];
  const float* dec_fnb = (const float*)d_in[44];
  const float* fcw = (const float*)d_in[45];
  const float* fcb = (const float*)d_in[46];

  // ---- workspace carve (floats)
  float* ws = (float*)d_ws;
  ll off = 0;
  auto take = [&](ll n) { float* p = ws + off; off += n; return p; };
  float* h      = take((ll)kBT * kD);
  float* xn     = take((ll)kBT * kD);
  float* g      = take((ll)kBT * kD);
  float* encb   = take((ll)kBT * kD);
  float* q      = take((ll)kBH * kT * kHS);
  float* k      = take((ll)kBH * kT * kHS);
  float* v      = take((ll)kBH * kT * kHS);
  float* attno  = take((ll)kBT * kD);
  float* scores = take((ll)kBH * kT * kT);
  float* ffmid  = take((ll)kBT * kFF);
  float* rowls  = take((ll)kBT);

  float* logits = (float*)d_out;
  float* lossp  = logits + (ll)kBT * kV;

  const BOffs z0{1, 0, 1, 0};
  const BOffs oaX{kH, (ll)kT * kD, 1, 0};        // activation rows per batch b
  const BOffs obW{1, 0, kH, (ll)kD * kHS};       // per-head weight
  const BOffs ocQ{1, (ll)kT * kHS, 1, 0};        // q/k/v per (b,h)
  const BOffs oaS{1, (ll)kT * kHS, 1, 0};
  const BOffs ocS{1, (ll)kT * kT, 1, 0};
  const BOffs oaP{1, (ll)kT * kT, 1, 0};
  const BOffs ocO{kH, (ll)kT * kD, kH, (ll)kHS}; // head-concat scatter

  auto GEMM = [&](const float* A, const float* Bp, const float* bias,
                  const float* Res, float* C, int M, int N, int K, int lda,
                  int ldb, int ldc, int batch, BOffs oa, BOffs ob, BOffs oc,
                  int tB, int relu, float alpha) {
    dim3 grid(N / 64, M / 64, batch);
    gemm_bf16_wmma<<<grid, 128, 0, stream>>>(A, Bp, bias, Res, C, M, N, K,
                                             lda, ldb, ldc, oa, ob, oc, tB,
                                             relu, alpha);
  };
  auto LN = [&](const float* in, const float* gg, const float* bb,
                float* out) {
    ln_kernel<<<kBT, 128, 0, stream>>>(in, gg, bb, out, kD);
  };
  auto ATTN = [&](const float* xq, const float* xkv, const float* wq,
                  const float* wk, const float* wv, const float* wo,
                  const float* bo, const int* km, int causal, float* res_io) {
    GEMM(xq,  wq, nullptr, nullptr, q, kT, kHS, kD, kD, kHS, kHS, kBH,
         oaX, obW, ocQ, 0, 0, 1.f);
    GEMM(xkv, wk, nullptr, nullptr, k, kT, kHS, kD, kD, kHS, kHS, kBH,
         oaX, obW, ocQ, 0, 0, 1.f);
    GEMM(xkv, wv, nullptr, nullptr, v, kT, kHS, kD, kD, kHS, kHS, kBH,
         oaX, obW, ocQ, 0, 0, 1.f);
    GEMM(q, k, nullptr, nullptr, scores, kT, kT, kHS, kHS, kHS, kT, kBH,
         oaS, oaS, ocS, 1, 0, 0.125f);               // alpha = HS^-0.5
    softmax_mask<<<kBH * kT, kT, 0, stream>>>(scores, km, causal, kT, kH);
    GEMM(scores, v, nullptr, nullptr, attno, kT, kHS, kT, kT, kHS, kD, kBH,
         oaP, oaS, ocO, 0, 0, 1.f);
    GEMM(attno, wo, bo, res_io, res_io, kBT, kD, kD, kD, kD, kD, 1,
         z0, z0, z0, 0, 0, 1.f);                     // += residual
  };

  const ll sW = (ll)kH * kD * kHS, sDD = (ll)kD * kD, sDF = (ll)kD * kFF;

  // ================= encoder =================
  {
    const int blocks = (int)(((ll)kBT * kD + 255) / 256);
    embed_kernel<<<blocks, 256, 0, stream>>>(x, enc_tok, enc_pos, h, kT, kD,
                                             (ll)kBT * kD);
  }
  for (int i = 0; i < kL; ++i) {
    LN(h, e_ln1g + i * kD, e_ln1b + i * kD, xn);
    ATTN(xn, xn, e_wq + i * sW, e_wk + i * sW, e_wv + i * sW,
         e_wo + i * sDD, e_bo + i * kD, mask, 0, h);
    LN(h, e_ln2g + i * kD, e_ln2b + i * kD, xn);
    GEMM(xn, e_w1 + i * sDF, e_b1 + i * kFF, nullptr, ffmid, kBT, kFF, kD,
         kD, kFF, kFF, 1, z0, z0, z0, 0, 1, 1.f);    // relu
    GEMM(ffmid, e_w2 + i * sDF, e_b2 + i * kD, h, h, kBT, kD, kFF, kFF, kD,
         kD, 1, z0, z0, z0, 0, 0, 1.f);
  }
  LN(h, enc_fng, enc_fnb, encb);

  // ================= decoder =================
  {
    const int blocks = (int)(((ll)kBT * kD + 255) / 256);
    embed_kernel<<<blocks, 256, 0, stream>>>(y, dec_tok, dec_pos, g, kT, kD,
                                             (ll)kBT * kD);
  }
  for (int i = 0; i < kL; ++i) {
    LN(g, d_ln1g + i * kD, d_ln1b + i * kD, xn);
    ATTN(xn, xn, d_sa_wq + i * sW, d_sa_wk + i * sW, d_sa_wv + i * sW,
         d_sa_wo + i * sDD, d_sa_bo + i * kD, tmask, 1, g);
    LN(g, d_ln2g + i * kD, d_ln2b + i * kD, g);      // in-place LN
    ATTN(g, encb, d_ca_wq + i * sW, d_ca_wk + i * sW, d_ca_wv + i * sW,
         d_ca_wo + i * sDD, d_ca_bo + i * kD, mask, 0, g);
    LN(g, d_ln3g + i * kD, d_ln3b + i * kD, g);      // in-place LN
    GEMM(g, d_w1 + i * sDF, d_b1 + i * kFF, nullptr, ffmid, kBT, kFF, kD,
         kD, kFF, kFF, 1, z0, z0, z0, 0, 1, 1.f);
    GEMM(ffmid, d_w2 + i * sDF, d_b2 + i * kD, g, g, kBT, kD, kFF, kFF, kD,
         kD, 1, z0, z0, z0, 0, 0, 1.f);
  }
  LN(g, dec_fng, dec_fnb, xn);

  // ================= logits + loss =================
  GEMM(xn, fcw, fcb, nullptr, logits, kBT, kV, kD, kD, kV, kV, 1,
       z0, z0, z0, 0, 0, 1.f);
  loss_kernel<<<kBT, 256, 0, stream>>>(logits, y, rowls, kV);
  reduce_loss<<<1, 256, 0, stream>>>(rowls, lossp, kBT);
}
